// SAU_47691316855554
// MI455X (gfx1250) — matrix operand order repaired and checked by reading, not compiled, not run on hardware
//
#include <hip/hip_runtime.h>
#include <hip/hip_bf16.h>

typedef __bf16 bf16_t;
typedef __attribute__((ext_vector_type(16))) __bf16 v16bf;
typedef __attribute__((ext_vector_type(8)))  __bf16 v8bf;
typedef __attribute__((ext_vector_type(8)))  float  v8f;
typedef __attribute__((ext_vector_type(4)))  unsigned v4ui;
typedef __attribute__((ext_vector_type(4)))  int    v4i;
typedef __attribute__((ext_vector_type(8)))  int    v8i;

#define BATCH 8
#define CH    256
#define NSP   4096
#define DIN   64     // inter = C/4
#define KT    64     // keys per tile in flash loop

#if defined(__has_builtin)
#  if __has_builtin(__builtin_amdgcn_tensor_load_to_lds)
#    define HAVE_TDM 1
#  endif
#endif
#ifndef HAVE_TDM
#  define HAVE_TDM 0
#endif

__device__ __forceinline__ v8f wmma_bf16(v16bf a, v16bf b, v8f c) {
    return __builtin_amdgcn_wmma_f32_16x16x32_bf16(false, a, false, b, (short)0, c, false, false);
}

// B-fragment: k = 16*hi + e is one contiguous 32B run.
__device__ __forceinline__ v16bf ld_bfrag32(const bf16_t* p) {
    v8bf lo = *(const v8bf*)(p);
    v8bf hi = *(const v8bf*)(p + 8);
    return __builtin_shufflevector(lo, hi, 0,1,2,3,4,5,6,7,8,9,10,11,12,13,14,15);
}

// A-fragment: k = 16*hi + e, one contiguous 32B run.
__device__ __forceinline__ v16bf ld_afrag(const bf16_t* row, int koff, int hi) {
    v8bf lo = *(const v8bf*)(row + koff + hi * 16);
    v8bf h2 = *(const v8bf*)(row + koff + hi * 16 + 8);
    return __builtin_shufflevector(lo, h2, 0,1,2,3,4,5,6,7,8,9,10,11,12,13,14,15);
}

// ---- data movement: TDM descriptors (preferred) or per-lane async (fallback)
#if HAVE_TDM
// D# group1 for a 2-D bf16 tile (ISA 08 §8.4): data_size=1 (2B), LDS padding
// of 4 dwords (code 3) after every (2<<padcode) dwords.
__device__ __forceinline__ v8i tdm_g1(unsigned td0, unsigned td1,
                                      unsigned tile0, unsigned tile1,
                                      unsigned stride0, unsigned padcode) {
    v8i g1;
    g1[0] = (int)((1u << 16) | (1u << 20) | (padcode << 22) | (3u << 25));
    g1[1] = (int)((td0 & 0xFFFFu) << 16);
    g1[2] = (int)(((td0 >> 16) & 0xFFFFu) | ((td1 & 0xFFFFu) << 16));
    g1[3] = (int)(((td1 >> 16) & 0xFFFFu) | (tile0 << 16));
    g1[4] = (int)(tile1 & 0xFFFFu);
    g1[5] = (int)stride0;          // tensor_dim0_stride[31:0]
    g1[6] = 0;
    g1[7] = 0;
    return g1;
}
__device__ __forceinline__ void tdm_load_2d(unsigned lds_addr, const void* g, v8i g1) {
    const unsigned long long ga = (unsigned long long)g;
    v4ui g0 = { 1u, lds_addr, (unsigned)ga,
                (unsigned)((ga >> 32) & 0x01FFFFFFull) | 0x80000000u }; // type=2
    v4i z4 = {0,0,0,0};
#if __clang_major__ >= 23
    v8i z8 = {0,0,0,0,0,0,0,0};
    __builtin_amdgcn_tensor_load_to_lds(g0, g1, z4, z4, z8, 0);
#else
    __builtin_amdgcn_tensor_load_to_lds(g0, g1, z4, z4, 0);
#endif
}
__device__ __forceinline__ unsigned lds_off(const void* p) {
    return (unsigned)(unsigned long long)p;
}
#else
__device__ __forceinline__ void async_load16(const bf16_t* g, bf16_t* l) {
    unsigned loff = (unsigned)(unsigned long long)(void*)l;
    asm volatile("global_load_async_to_lds_b128 %0, %1, off"
                 :: "v"(loff), "v"(g) : "memory");
}
__device__ __forceinline__ void wait_async0() {
    asm volatile("s_wait_asynccnt 0x0" ::: "memory");
}
#endif

__device__ __forceinline__ void wait_tiles() {
#if HAVE_TDM
    __builtin_amdgcn_s_wait_tensorcnt(0);
#else
    wait_async0();
#endif
}

// ---------------------------------------------------------------------------
// Kernel 0: one-time prep. Blocks [0, 2048): transpose x -> XT (B, N, C) bf16.
// Blocks [2048, 2144): weights -> bf16 (wq pre-scaled by 1/8), [wq|wk|wv].
// ---------------------------------------------------------------------------
__global__ __launch_bounds__(256) void prep_kernel(
    const float* __restrict__ x,
    const float* __restrict__ wq, const float* __restrict__ wk,
    const float* __restrict__ wv,
    bf16_t* __restrict__ XT, bf16_t* __restrict__ Wbf)
{
    const int t = threadIdx.x;
    if (blockIdx.x < 2048) {
        __shared__ bf16_t Tsh[64 * 65];          // [c][n], odd stride
        const int bid = blockIdx.x;
        const int b  = bid >> 8;
        const int c0 = ((bid >> 6) & 3) * 64;
        const int n0 = (bid & 63) * 64;
        const float* xb = x + ((size_t)b * CH + c0) * NSP + n0;
        {   // read coalesced over n
            const int j = t & 63, ci = t >> 6;
            for (int i = 0; i < 16; ++i) {
                const int c = i * 4 + ci;
                Tsh[c * 65 + j] = (bf16_t)xb[(size_t)c * NSP + j];
            }
        }
        __syncthreads();
        {   // write coalesced over c
            const int ic = t & 63, jn = t >> 6;
            for (int i = 0; i < 16; ++i) {
                const int n = i * 4 + jn;
                XT[((size_t)b * NSP + n0 + n) * CH + c0 + ic] = Tsh[ic * 65 + n];
            }
        }
    } else {
        const int wb = blockIdx.x - 2048;        // 96 blocks x 1024 elems
        const int base = wb * 1024 + t * 4;
        #pragma unroll
        for (int u = 0; u < 4; ++u) {
            const int i = base + u;
            float v;
            if (i < 16384)      v = wq[i] * 0.125f;
            else if (i < 32768) v = wk[i - 16384];
            else                v = wv[i - 32768];
            Wbf[i] = (bf16_t)v;
        }
    }
}

// ---------------------------------------------------------------------------
// Kernel 1: Q/K/V projections (all-bf16). Qb (B,N,64) | Kb (B,N,64) | Vb (B,C,N)
// X tile staged into LDS [n][c+pad] by one TDM descriptor (or async fallback).
// ---------------------------------------------------------------------------
__global__ __launch_bounds__(256) void proj_kernel(
    const bf16_t* __restrict__ XT, const bf16_t* __restrict__ Wbf,
    const float* __restrict__ bq, const float* __restrict__ bk,
    const float* __restrict__ bv,
    bf16_t* __restrict__ Qb, bf16_t* __restrict__ Kb, bf16_t* __restrict__ Vb)
{
    __shared__ bf16_t Xsh[64 * 264];   // [n][c + 8 pad] = 33 KB

    const int t  = threadIdx.x;
    const int b  = blockIdx.x >> 6;
    const int n0 = (blockIdx.x & 63) * 64;
    const int wid = t >> 5;

    const bf16_t* xrow = XT + ((size_t)b * NSP + n0) * CH;
#if HAVE_TDM
    if (wid == 0)   // 64 rows x 512B; pad after 128 dwords (code 6)
        tdm_load_2d(lds_off(Xsh), xrow, tdm_g1(CH, NSP, CH, 64, CH, 6));
#else
    #pragma unroll
    for (int i = 0; i < 8; ++i) {
        const int idx = i * 256 + t;
        const int n = idx >> 5, j = idx & 31;
        async_load16(xrow + (size_t)n * CH + j * 8, Xsh + n * 264 + j * 8);
    }
#endif
    wait_tiles();
    __syncthreads();

    const int lane = t & 31;
    const int hi   = lane >> 4;
    const int ln   = lane & 15;
    const int strip = (wid & 3) * 16;
    const int half  = wid >> 2;

    v16bf bx[8];
    #pragma unroll
    for (int kk = 0; kk < 8; ++kk)
        bx[kk] = ld_bfrag32(Xsh + (strip + ln) * 264 + kk * 32 + hi * 16);

    // 24 output subtiles: idx 0..3 = Q, 4..7 = K, 8..23 = V. Split 12/12.
    for (int ii = 0; ii < 12; ++ii) {
        const int idx = half * 12 + ii;
        const bf16_t* w;  const float* bias;  bf16_t* outp;  int ob;
        float bscale;  size_t obase;  unsigned ostride;
        const int n = n0 + strip + ln;
        if (idx < 4) {
            w = Wbf;         bias = bq; outp = Qb; ob = idx;     bscale = 0.125f;
            obase = ((size_t)b * NSP + n) * 64;  ostride = 1;
        } else if (idx < 8) {
            w = Wbf + 16384; bias = bk; outp = Kb; ob = idx - 4; bscale = 1.0f;
            obase = ((size_t)b * NSP + n) * 64;  ostride = 1;
        } else {
            w = Wbf + 32768; bias = bv; outp = Vb; ob = idx - 8; bscale = 1.0f;
            obase = (size_t)b * CH * NSP + n;    ostride = NSP;
        }

        const bf16_t* wrow = w + (size_t)(ob * 16 + ln) * CH;
        v8f acc = {0.f,0.f,0.f,0.f,0.f,0.f,0.f,0.f};
        #pragma unroll
        for (int kk = 0; kk < 8; ++kk) {
            v16bf af = ld_afrag(wrow, kk * 32, hi);   // A = weights: m=o, k=c
            acc = wmma_bf16(af, bx[kk], acc);
        }
        // bias for rows o = ob*16 + 8*hi + (0..7): two float4 loads
        const float4 bv0 = *(const float4*)(bias + ob * 16 + 8 * hi);
        const float4 bv1 = *(const float4*)(bias + ob * 16 + 8 * hi + 4);
        const float bb[8] = {bv0.x, bv0.y, bv0.z, bv0.w, bv1.x, bv1.y, bv1.z, bv1.w};
        #pragma unroll
        for (int r = 0; r < 8; ++r) {
            const unsigned o = (unsigned)(ob * 16 + r + 8 * hi);
            outp[obase + (size_t)o * ostride] = (bf16_t)fmaf(bb[r], bscale, acc[r]);
        }
    }
}

// ---------------------------------------------------------------------------
// Kernel 2: flash attention, double-buffered TDM pipeline.
// Per tile: wait(tensorcnt) -> barrier -> issue DMA(kt+1) -> compute(kt).
// ---------------------------------------------------------------------------
__global__ __launch_bounds__(256) void attn_kernel(
    const bf16_t* __restrict__ Qb, const bf16_t* __restrict__ Kb,
    const bf16_t* __restrict__ Vb, float* __restrict__ out)
{
    __shared__ bf16_t Ksh[2][KT * 72];    // 2 x  9 KB  [key][d + pad]
    __shared__ bf16_t Vsh[2][CH * 72];    // 2 x 36 KB  [c][key + pad]

    const int t  = threadIdx.x;
    const int b  = blockIdx.x >> 6;
    const int q0 = (blockIdx.x & 63) * 64;

    const int lane  = t & 31;
    const int wid   = t >> 5;
    const int hi    = lane >> 4;
    const int ln    = lane & 15;
    const int qs    = q0 + (wid >> 1) * 16;  // query strip base
    const int chalf = wid & 1;               // channel half 0/1

    const bf16_t* Kbase = Kb + (size_t)b * NSP * DIN;
    const bf16_t* Vbase = Vb + (size_t)b * CH * NSP;   // (C, N)

    // ---- persistent Q B-fragments
    v16bf qf[2];
    {
        const bf16_t* qrow = Qb + ((size_t)b * NSP + qs + ln) * DIN;
        qf[0] = ld_bfrag32(qrow + hi * 16);
        qf[1] = ld_bfrag32(qrow + 32 + hi * 16);
    }

    v8f accO[8];
    #pragma unroll
    for (int i = 0; i < 8; ++i) accO[i] = (v8f){0.f,0.f,0.f,0.f,0.f,0.f,0.f,0.f};
    float m_run = -3.0e30f, l_run = 0.0f;
    const float L2E = 1.4426950408889634f;

    auto issue_tile = [&](int kt, int bu) {
#if HAVE_TDM
        if (wid == 0) {   // TDM ignores EXEC; issue once per workgroup
            tdm_load_2d(lds_off(&Ksh[bu][0]), Kbase + (size_t)kt * KT * DIN,
                        tdm_g1(DIN, NSP, DIN, KT, DIN, 4));
            tdm_load_2d(lds_off(&Vsh[bu][0]), Vbase + (size_t)kt * KT,
                        tdm_g1(NSP, CH, KT, CH, NSP, 4));
        }
#else
        #pragma unroll
        for (int i = 0; i < 2; ++i) {             // K: 512 16B chunks
            const int idx = i * 256 + t;
            const int key = idx >> 3, j = idx & 7;
            async_load16(Kbase + (size_t)(kt * KT + key) * DIN + j * 8,
                         &Ksh[bu][key * 72 + j * 8]);
        }
        #pragma unroll
        for (int i = 0; i < 8; ++i) {             // V: 2048 16B chunks
            const int idx = i * 256 + t;
            const int c = idx >> 3, j = idx & 7;
            async_load16(Vbase + (size_t)c * NSP + kt * KT + j * 8,
                         &Vsh[bu][c * 72 + j * 8]);
        }
#endif
    };

    issue_tile(0, 0);   // prologue

    for (int kt = 0; kt < NSP / KT; ++kt) {
        const int bu = kt & 1;
        wait_tiles();                  // tile kt landed (TENSORcnt/ASYNCcnt)
        __syncthreads();               // all waves see it; prev compute done
        if (kt + 1 < NSP / KT) issue_tile(kt + 1, bu ^ 1);   // overlap DMA

        // ---- S^T = K * Q^T : 4 subtiles of 16 keys (2 WMMA each)
        v8f Sf[4];
        #pragma unroll
        for (int s = 0; s < 4; ++s) {
            const bf16_t* krow = &Ksh[bu][(s * 16 + ln) * 72];
            v16bf a0 = ld_afrag(krow, 0,  hi);
            v16bf a1 = ld_afrag(krow, 32, hi);
            v8f sa = {0.f,0.f,0.f,0.f,0.f,0.f,0.f,0.f};
            sa = wmma_bf16(a0, qf[0], sa);
            sa = wmma_bf16(a1, qf[1], sa);
            Sf[s] = sa;   // row key = s*16 + r + 8*hi, col query = ln
        }

        // ---- online softmax (per-lane regs + one lane-16 fold)
        float tmax = -3.0e30f;
        #pragma unroll
        for (int s = 0; s < 4; ++s)
            #pragma unroll
            for (int r = 0; r < 8; ++r) tmax = fmaxf(tmax, Sf[s][r]);
        tmax = fmaxf(tmax, __shfl_xor(tmax, 16, 32));
        const float newm = fmaxf(m_run, tmax);
        const float scale_old = exp2f((m_run - newm) * L2E);
        float tsum = 0.f;
        #pragma unroll
        for (int s = 0; s < 4; ++s)
            #pragma unroll
            for (int r = 0; r < 8; ++r) {
                const float p = exp2f((Sf[s][r] - newm) * L2E);
                Sf[s][r] = p;
                tsum += p;
            }
        tsum += __shfl_xor(tsum, 16, 32);
        l_run = l_run * scale_old + tsum;
        m_run = newm;
        #pragma unroll
        for (int i = 0; i < 8; ++i) accO[i] = accO[i] * scale_old;

        // ---- repack P (keys x queries) into B-frags
        v16bf pb[2];
        #pragma unroll
        for (int g = 0; g < 2; ++g) {
            #pragma unroll
            for (int r = 0; r < 8; ++r) {
                const float send = hi ? Sf[2 * g][r] : Sf[2 * g + 1][r];
                const float recv = __shfl_xor(send, 16, 32);
                const float own  = hi ? Sf[2 * g + 1][r] : Sf[2 * g][r];
                pb[g][r]     = (bf16_t)(hi ? recv : own);   // key row r
                pb[g][r + 8] = (bf16_t)(hi ? own  : recv);  // key row r+8
            }
        }

        // ---- O^T += V^T * P^T : 8 channel subtiles x 2 key groups
        #pragma unroll
        for (int cs = 0; cs < 8; ++cs) {
            const bf16_t* vrow = &Vsh[bu][(chalf * 128 + cs * 16 + ln) * 72];
            #pragma unroll
            for (int g = 0; g < 2; ++g) {
                v16bf vf = ld_afrag(vrow, 32 * g, hi);   // A = V^T: m=c, k=key
                accO[cs] = wmma_bf16(vf, pb[g], accO[cs]);
            }
        }
    }

    // ---- epilogue: divide by softmax denominator, scatter to (B, C, N)
    const float inv = 1.0f / l_run;
    float* outb = out + (size_t)b * CH * NSP;
    #pragma unroll
    for (int cs = 0; cs < 8; ++cs) {
        #pragma unroll
        for (int r = 0; r < 8; ++r) {
            const int c = chalf * 128 + cs * 16 + r + 8 * hi;
            outb[(size_t)c * NSP + qs + ln] = accO[cs][r] * inv;
        }
    }
}

// ---------------------------------------------------------------------------
extern "C" void kernel_launch(void* const* d_in, const int* in_sizes, int n_in,
                              void* d_out, int out_size, void* d_ws, size_t ws_size,
                              hipStream_t stream) {
    const float* x  = (const float*)d_in[0];
    const float* wq = (const float*)d_in[1];
    const float* bq = (const float*)d_in[2];
    const float* wk = (const float*)d_in[3];
    const float* bk = (const float*)d_in[4];
    const float* wv = (const float*)d_in[5];
    const float* bv = (const float*)d_in[6];
    float* out = (float*)d_out;

    // workspace (bf16): Qb 4MiB | Kb 4MiB | Vb 16MiB | XT 16MiB | Wbf 192KiB
    bf16_t* Qb  = (bf16_t*)d_ws;
    bf16_t* Kb  = Qb + (size_t)BATCH * NSP * DIN;
    bf16_t* Vb  = Kb + (size_t)BATCH * NSP * DIN;
    bf16_t* XT  = Vb + (size_t)BATCH * CH * NSP;
    bf16_t* Wbf = XT + (size_t)BATCH * NSP * CH;

    prep_kernel<<<2048 + 96, 256, 0, stream>>>(x, wq, wk, wv, XT, Wbf);
    proj_kernel<<<BATCH * (NSP / 64), 256, 0, stream>>>(XT, Wbf, bq, bk, bv, Qb, Kb, Vb);
    attn_kernel<<<BATCH * (NSP / 64), 256, 0, stream>>>(Qb, Kb, Vb, out);
}
